// _MultiheadNonlocalNd_63642825392174
// MI455X (gfx1250) — matrix operand order repaired and checked by reading, not compiled, not run on hardware
//
#include <hip/hip_runtime.h>
#include <hip/hip_bf16.h>

// ---------------------------------------------------------------------------
// MultiheadNonlocal attention for MI455X (gfx1250), wave32 + WMMA f16 +
// async global->LDS staging (double-buffered) in the flash-attention loop.
// B=4, C=256, H=W=64 (N=4096), heads=2, hidden=64, scale = C^-0.5 = 0.0625
// ---------------------------------------------------------------------------

typedef __attribute__((ext_vector_type(16))) _Float16 v16h;
typedef __attribute__((ext_vector_type(8)))  float    v8f;
typedef int v4i_vs __attribute__((vector_size(16)));   // matches builtin proto

#define BB   4
#define CC   256
#define NN   4096
#define HH   2
#define HC   64
#define BH   (BB*HH)

#if __has_builtin(__builtin_amdgcn_global_load_async_to_lds_b128) && \
    __has_builtin(__builtin_amdgcn_s_wait_asynccnt)
#define USE_ASYNC 1
#else
#define USE_ASYNC 0
#endif

// --- WMMA helpers -----------------------------------------------------------

__device__ __forceinline__ v8f wmma16(v16h a, v16h b, v8f c) {
  // D = A(16x32 f16) * B(32x16 f16) + C(16x16 f32)
  return __builtin_amdgcn_wmma_f32_16x16x32_f16(
      /*neg_a=*/false, a, /*neg_b=*/false, b,
      /*c_mod=*/(short)0, c, /*reuse_a=*/false, /*reuse_b=*/false);
}

// A fragment (16x32, f16) from row-major memory, row stride ld (halves).
// Lane layout (ISA 7.12.2): row = lane&15; halves 0..7 = K koff..koff+7,
// halves 8..15 = K koff+16..koff+23, koff = (lane<16)?0:8.
__device__ __forceinline__ v16h frag_a(const _Float16* base, int ld, int lane) {
  int row  = lane & 15;
  int koff = (lane & 16) ? 8 : 0;
  const _Float16* p = base + row * ld + koff;
  union { v16h v; uint4 q[2]; } u;
  u.q[0] = *(const uint4*)(p);
  u.q[1] = *(const uint4*)(p + 16);
  return u.v;
}

// B fragment (32x16, f16) from memory laid out [col][k] (K contiguous per
// column), column stride ld. Lane: col = lane&15; halves 0..7 = K kb..kb+7,
// halves 8..15 = K kb+16..kb+23, kb = (lane<16)?0:8.
__device__ __forceinline__ v16h frag_b(const _Float16* base, int ld, int lane) {
  int col = lane & 15;
  int kb  = (lane & 16) ? 8 : 0;
  const _Float16* p = base + col * ld + kb;
  union { v16h v; uint4 q[2]; } u;
  u.q[0] = *(const uint4*)(p);
  u.q[1] = *(const uint4*)(p + 16);
  return u.v;
}

// 16-byte global -> LDS copy: async (ASYNCcnt) on gfx1250, sync fallback.
__device__ __forceinline__ void copy16(const _Float16* g, _Float16* l) {
#if USE_ASYNC
  __builtin_amdgcn_global_load_async_to_lds_b128(
      (v4i_vs*)g, (v4i_vs*)l, 0, 0);
#else
  *(uint4*)l = *(const uint4*)g;
#endif
}

// --- 1) x [B,C,N] f32  ->  xT [B,N,C] f16 (LDS tile transpose) -------------

__global__ __launch_bounds__(256) void pack_x(const float* __restrict__ x,
                                              _Float16* __restrict__ xT) {
  __shared__ _Float16 t[64][72];            // [n][c], padded vs bank conflicts
  int b  = blockIdx.z;
  int nt = blockIdx.x * 64;
  int ct = blockIdx.y * 64;
  for (int i = 0; i < 16; ++i) {
    int idx = i * 256 + threadIdx.x;        // 0..4095
    int n = idx & 63, c = idx >> 6;         // reads coalesced in n
    t[n][c] = (_Float16)x[((size_t)(b * CC + ct + c)) * NN + nt + n];
  }
  __syncthreads();
  for (int i = 0; i < 16; ++i) {
    int idx = i * 256 + threadIdx.x;
    int c = idx & 63, n = idx >> 6;         // writes coalesced in c
    xT[((size_t)(b * NN + nt + n)) * CC + ct + c] = t[n][c];
  }
}

// --- 2) weights -> f16 (embed_w as-is, out_w transposed) --------------------

__global__ __launch_bounds__(256) void pack_w(const float* __restrict__ ew,
                                              const float* __restrict__ ow,
                                              _Float16* __restrict__ wqkv,
                                              _Float16* __restrict__ w2T) {
  int i = blockIdx.x * 256 + threadIdx.x;
  if (i < HH * 3 * HC * CC) wqkv[i] = (_Float16)ew[i];
  if (i < CC * 2 * HC) {                    // w2T[c][d] = out_w[d][c]
    int c = i >> 7, d = i & 127;
    w2T[i] = (_Float16)ow[d * CC + c];
  }
}

// --- 3) qT,kT [bh][n][64] = xT * Wqk^T --------------------------------------
// One wave: 16 query rows x 128 cols (q:0..63, k:64..127), K=C=256.

__global__ __launch_bounds__(256) void qk_proj(const _Float16* __restrict__ xT,
                                               const _Float16* __restrict__ wqkv,
                                               const float* __restrict__ eb,
                                               _Float16* __restrict__ qT,
                                               _Float16* __restrict__ kT) {
  int lane = threadIdx.x & 31, widx = threadIdx.x >> 5;
  int W  = blockIdx.x * 8 + widx;           // 2048 waves total
  int bh = W >> 8;                          // 0..7
  int b = bh >> 1, h = bh & 1;
  int n0 = (W & 255) << 4;
  const _Float16* A  = xT + ((size_t)(b * NN + n0)) * CC;
  const _Float16* Bw = wqkv + (size_t)h * 3 * HC * CC;   // rows d=0..127 (q,k)
  v8f acc[8] = {};
  for (int kc = 0; kc < CC; kc += 32) {
    v16h a = frag_a(A + kc, CC, lane);
    v16h bf[8];
    for (int f = 0; f < 8; ++f)
      bf[f] = frag_b(Bw + (size_t)(f * 16) * CC + kc, CC, lane);
    for (int f = 0; f < 8; ++f)
      acc[f] = wmma16(a, bf[f], acc[f]);
  }
  int colL = lane & 15, hi = lane >> 4;
  for (int f = 0; f < 8; ++f) {
    int d = f * 16 + colL;
    float bias = eb[h * 3 * HC + d];
    _Float16* dst = (d < HC) ? qT : kT;
    int dd = d & (HC - 1);
    for (int r = 0; r < 8; ++r) {
      int n = n0 + r + 8 * hi;
      dst[((size_t)(bh * NN + n)) * HC + dd] = (_Float16)(acc[f][r] + bias);
    }
  }
}

// --- 4) v [bh][d][n] = Wv * x ----------------------------------------------
// One wave: 16 d-rows x 64 n-cols, K=C=256.

__global__ __launch_bounds__(256) void v_proj(const _Float16* __restrict__ xT,
                                              const _Float16* __restrict__ wqkv,
                                              const float* __restrict__ eb,
                                              _Float16* __restrict__ vv) {
  int lane = threadIdx.x & 31, widx = threadIdx.x >> 5;
  int W  = blockIdx.x * 8 + widx;           // 2048 waves
  int bh = W >> 8;
  int rem = W & 255;
  int d0 = (rem >> 6) * 16;                 // 0,16,32,48
  int n0 = (rem & 63) * 64;
  int b = bh >> 1, h = bh & 1;
  const _Float16* Aw = wqkv + ((size_t)(h * 3 * HC + 2 * HC + d0)) * CC;
  const _Float16* Bx = xT + ((size_t)(b * NN + n0)) * CC;
  v8f acc[4] = {};
  for (int kc = 0; kc < CC; kc += 32) {
    v16h a = frag_a(Aw + kc, CC, lane);
    v16h bf[4];
    for (int f = 0; f < 4; ++f)
      bf[f] = frag_b(Bx + (size_t)(f * 16) * CC + kc, CC, lane);
    for (int f = 0; f < 4; ++f)
      acc[f] = wmma16(a, bf[f], acc[f]);
  }
  int colL = lane & 15, hi = lane >> 4;
  for (int f = 0; f < 4; ++f) {
    int n = n0 + f * 16 + colL;
    for (int r = 0; r < 8; ++r) {
      int d = d0 + r + 8 * hi;
      float bias = eb[h * 3 * HC + 2 * HC + d];
      vv[((size_t)(bh * HC + d)) * NN + n] = (_Float16)(acc[f][r] + bias);
    }
  }
}

// --- 5) flash attention: oT[b][n][h*64+d] -----------------------------------
// Block = 4 waves / 64 query rows; K & V tiles (64x64 f16 each) staged into
// LDS with double-buffered async global->LDS copies; wave owns 16 query rows.

__global__ __launch_bounds__(128) void flash_attn(const _Float16* __restrict__ qT,
                                                  const _Float16* __restrict__ kT,
                                                  const _Float16* __restrict__ vv,
                                                  _Float16* __restrict__ oT) {
  __shared__ __align__(16) _Float16 Kbuf[2][64 * 64];  // [n][c] tile, 8 KB each
  __shared__ __align__(16) _Float16 Vbuf[2][64 * 64];  // [d][n] tile, 8 KB each
  __shared__ __align__(16) _Float16 P[4][16][64];      // per-wave softmax tile
  const int T = NN / 64;                    // 64 key tiles
  int tid = threadIdx.x;
  int lane = tid & 31, w = tid >> 5;
  int bh = blockIdx.y;
  int b = bh >> 1, h = bh & 1;
  int q0 = blockIdx.x * 64 + w * 16;
  int colL = lane & 15, hi = lane >> 4;

  const _Float16* Kg = kT + (size_t)bh * NN * HC;   // [n][c], rows contiguous
  const _Float16* Vg = vv + (size_t)bh * HC * NN;   // [d][n], row stride NN

  // Stage one 64-wide key tile (K: 8KB contiguous, V: 64 rows x 128B).
  auto stage = [&](int kt, int buf) {
    int n0 = kt * 64;
    for (int j = 0; j < 4; ++j) {
      int i = j * 128 + tid;                          // 0..511 16B chunks
      copy16(Kg + (size_t)n0 * HC + i * 8, &Kbuf[buf][i * 8]);
    }
    for (int j = 0; j < 4; ++j) {
      int i = j * 128 + tid;
      int d = i >> 3, off = (i & 7) * 8;
      copy16(Vg + (size_t)d * NN + n0 + off, &Vbuf[buf][d * 64 + off]);
    }
  };

  const _Float16* Qb = qT + ((size_t)(bh * NN + q0)) * HC;
  v16h aq0 = frag_a(Qb, HC, lane);          // K = c 0..31
  v16h aq1 = frag_a(Qb + 32, HC, lane);     // K = c 32..63

  v8f o[4] = {};
  float mst[8], lst[8];
  for (int r = 0; r < 8; ++r) { mst[r] = -1e30f; lst[r] = 0.f; }
  const float scale = 0.0625f;              // 256^-0.5

  stage(0, 0);
  for (int kt = 0; kt < T; ++kt) {
    int cur = kt & 1;
    if (kt + 1 < T) stage(kt + 1, cur ^ 1);
    if (kt + 2 < T)                          // global_prefetch_b8 of tile kt+2
      __builtin_prefetch(Kg + (size_t)(kt + 2) * 64 * HC, 0, 3);
#if USE_ASYNC
    if (kt + 1 < T) __builtin_amdgcn_s_wait_asynccnt(8);  // tile kt complete
    else            __builtin_amdgcn_s_wait_asynccnt(0);
#endif
    __syncthreads();                        // all waves' tile data visible
    const _Float16* Kt = &Kbuf[cur][0];
    const _Float16* Vt = &Vbuf[cur][0];

    // S tile 16x64 = Q * K^T  (B-fragments from LDS, ld = 64)
    v8f s[4] = {};
    for (int f = 0; f < 4; ++f) {
      s[f] = wmma16(aq0, frag_b(Kt + f * 16 * 64, 64, lane), s[f]);
      s[f] = wmma16(aq1, frag_b(Kt + f * 16 * 64 + 32, 64, lane), s[f]);
    }
    for (int f = 0; f < 4; ++f) s[f] *= scale;

    // online softmax per row (rows live across a 16-lane half-wave)
    for (int r = 0; r < 8; ++r) {
      float mx = fmaxf(fmaxf(s[0][r], s[1][r]), fmaxf(s[2][r], s[3][r]));
      for (int off = 1; off < 16; off <<= 1)
        mx = fmaxf(mx, __shfl_xor(mx, off, 32));
      float nm   = fmaxf(mst[r], mx);
      float corr = __expf(mst[r] - nm);
      float rs = 0.f;
      int m = r + 8 * hi;
      for (int f = 0; f < 4; ++f) {
        float p = __expf(s[f][r] - nm);
        rs += p;
        P[w][m][f * 16 + colL] = (_Float16)p;
        o[f][r] *= corr;
      }
      for (int off = 1; off < 16; off <<= 1)
        rs += __shfl_xor(rs, off, 32);
      lst[r] = lst[r] * corr + rs;
      mst[r] = nm;
    }
    __syncthreads();                        // P visible as A-fragments
    v16h ap0 = frag_a(&P[w][0][0], 64, lane);
    v16h ap1 = frag_a(&P[w][0][0] + 32, 64, lane);
    for (int f = 0; f < 4; ++f) {
      o[f] = wmma16(ap0, frag_b(Vt + f * 16 * 64, 64, lane), o[f]);
      o[f] = wmma16(ap1, frag_b(Vt + f * 16 * 64 + 32, 64, lane), o[f]);
    }
    __syncthreads();                        // done with Kbuf/Vbuf[cur] and P
  }

  for (int f = 0; f < 4; ++f) {
    int d = h * HC + f * 16 + colL;
    for (int r = 0; r < 8; ++r) {
      int n = q0 + r + 8 * hi;
      oT[((size_t)(b * NN + n)) * (HH * HC) + d] = (_Float16)(o[f][r] / lst[r]);
    }
  }
}

// --- 6) out[b][c][n] = W2^T * o + bias + x ----------------------------------
// One wave: 16 c-rows x 64 n-cols, K=128. Stores coalesced along n.

__global__ __launch_bounds__(256) void out_proj(const _Float16* __restrict__ oT,
                                                const _Float16* __restrict__ w2T,
                                                const float* __restrict__ ob,
                                                const float* __restrict__ x,
                                                float* __restrict__ out) {
  int lane = threadIdx.x & 31, widx = threadIdx.x >> 5;
  int W = blockIdx.x * 8 + widx;            // 4096 waves
  int b = W >> 10;
  int rem = W & 1023;
  int c0 = (rem >> 6) * 16;
  int n0 = (rem & 63) * 64;
  const int KD = HH * HC;                   // 128
  const _Float16* A  = w2T + (size_t)c0 * KD;
  const _Float16* Bo = oT + ((size_t)(b * NN + n0)) * KD;
  v8f acc[4] = {};
  for (int kc = 0; kc < KD; kc += 32) {
    v16h a = frag_a(A + kc, KD, lane);
    v16h bf[4];
    for (int f = 0; f < 4; ++f)
      bf[f] = frag_b(Bo + (size_t)(f * 16) * KD + kc, KD, lane);
    for (int f = 0; f < 4; ++f)
      acc[f] = wmma16(a, bf[f], acc[f]);
  }
  int colL = lane & 15, hi = lane >> 4;
  for (int f = 0; f < 4; ++f) {
    int n = n0 + f * 16 + colL;
    for (int r = 0; r < 8; ++r) {
      int c = c0 + r + 8 * hi;
      size_t idx = ((size_t)(b * CC + c)) * NN + n;
      out[idx] = acc[f][r] + ob[c] + x[idx];
    }
  }
}

// --- host -------------------------------------------------------------------

extern "C" void kernel_launch(void* const* d_in, const int* in_sizes, int n_in,
                              void* d_out, int out_size, void* d_ws, size_t ws_size,
                              hipStream_t stream) {
  (void)in_sizes; (void)n_in; (void)out_size; (void)ws_size;
  const float* x  = (const float*)d_in[0];
  const float* ew = (const float*)d_in[1];
  const float* eb = (const float*)d_in[2];
  const float* ow = (const float*)d_in[3];
  const float* ob = (const float*)d_in[4];
  float* out = (float*)d_out;

  char* ws = (char*)d_ws;
  size_t o_xT   = 0;                                 // 4*4096*256 f16 = 8 MB
  size_t o_wqkv = o_xT + (size_t)BB * NN * CC * 2;   // 2*192*256 f16
  size_t o_w2T  = o_wqkv + (size_t)HH * 3 * HC * CC * 2;
  size_t o_qT   = o_w2T + (size_t)CC * 2 * HC * 2;   // 8*4096*64 f16 = 4 MB
  size_t o_kT   = o_qT + (size_t)BH * NN * HC * 2;
  size_t o_v    = o_kT + (size_t)BH * NN * HC * 2;
  size_t o_oT   = o_v  + (size_t)BH * NN * HC * 2;   // 4*4096*128 f16 = 4 MB
  _Float16* xT   = (_Float16*)(ws + o_xT);
  _Float16* wqkv = (_Float16*)(ws + o_wqkv);
  _Float16* w2T  = (_Float16*)(ws + o_w2T);
  _Float16* qT   = (_Float16*)(ws + o_qT);
  _Float16* kT   = (_Float16*)(ws + o_kT);
  _Float16* vv   = (_Float16*)(ws + o_v);
  _Float16* oT   = (_Float16*)(ws + o_oT);

  pack_x<<<dim3(NN / 64, CC / 64, BB), 256, 0, stream>>>(x, xT);
  pack_w<<<(HH * 3 * HC * CC + 255) / 256, 256, 0, stream>>>(ew, ow, wqkv, w2T);
  qk_proj<<<256, 256, 0, stream>>>(xT, wqkv, eb, qT, kT);
  v_proj<<<256, 256, 0, stream>>>(xT, wqkv, eb, vv);
  flash_attn<<<dim3(NN / 64, BH), 128, 0, stream>>>(qT, kT, vv, oT);
  out_proj<<<512, 256, 0, stream>>>(oT, w2T, ob, x, out);
}
